// RGGC_54082228191675
// MI455X (gfx1250) — matrix operand order repaired and checked by reading, not compiled, not run on hardware
//
#include <hip/hip_runtime.h>

#define NN 50000
#define NE 400000
#define DD 128
#define MTILES (NN / 16)   // 3125 exact

typedef __attribute__((ext_vector_type(16))) __bf16        v16bf;
typedef __attribute__((ext_vector_type(8)))  float         v8f;
typedef __attribute__((ext_vector_type(8)))  unsigned int  v8u;

union BF16x16 {
    v8u            u;
    v16bf          v;
    unsigned short s[16];
};

__device__ __forceinline__ unsigned short f2bf(float f) {
    unsigned u = __float_as_uint(f);
    u += 0x7FFFu + ((u >> 16) & 1u);      // round-to-nearest-even
    return (unsigned short)(u >> 16);
}
__device__ __forceinline__ float bf2f(unsigned short h) {
    return __uint_as_float(((unsigned)h) << 16);
}

// ---------------------------------------------------------------------------
// Pack 4 fp32 128x128 weights into hi/lo bf16 fragments laid out as
// WMMA B tiles (32x16) in contiguous k order, matching A: lane = (K>>4 within chunk)*16 + n,
// element e = K&15, 16 contiguous bf16 per lane (one 32B load in the GEMM).
// ---------------------------------------------------------------------------
__global__ void pack_weights(const float* __restrict__ W0, const float* __restrict__ W1,
                             const float* __restrict__ W2, const float* __restrict__ W3,
                             unsigned short* __restrict__ Bhi, unsigned short* __restrict__ Blo) {
    const int wi = blockIdx.x;
    const float* W = (wi == 0) ? W0 : (wi == 1) ? W1 : (wi == 2) ? W2 : W3;
    const int base = wi * (DD * DD);
    for (int i = threadIdx.x; i < DD * DD; i += blockDim.x) {
        const int k = i >> 7, n = i & 127;
        const float w = W[i];
        const unsigned short hi = f2bf(w);
        const unsigned short lo = f2bf(w - bf2f(hi));
        const int kt = k >> 5, kr = k & 31, nt = n >> 4, nc = n & 15;
        const int lane = ((kr >> 4) << 4) | nc;
        const int e    = kr & 15;
        const int idx  = base + (((kt * 8 + nt) * 32) + lane) * 16 + e;
        Bhi[idx] = hi;
        Blo[idx] = lo;
    }
}

// ---------------------------------------------------------------------------
// Fused 4-weight GEMM: each wave computes one 16-row node tile x 128 cols for
// Wk, Wq, Wv, Ws. A (activations) converted on the fly to bf16 hi/lo.
// 3-product bf16 split per k-chunk: Ahi*Bhi + Alo*Bhi + Ahi*Blo  (~fp32 acc).
// Ws path initializes the accumulator with the bias -> doubles as scatter init.
// ---------------------------------------------------------------------------
__global__ __launch_bounds__(256)
void gemm_kqvs(const float* __restrict__ X,
               const unsigned short* __restrict__ Bhi,
               const unsigned short* __restrict__ Blo,
               const float* __restrict__ bias,
               float* __restrict__ Ko, float* __restrict__ Qo,
               float* __restrict__ Vo, float* __restrict__ So,
               int relu) {
    const int gw = (blockIdx.x * blockDim.x + threadIdx.x) >> 5;
    if (gw >= MTILES) return;                      // wave-uniform guard
    const int lane = threadIdx.x & 31;
    const int m    = lane & 15;                    // row within tile (A), col within tile (C)
    const int h    = lane >> 4;                    // K-half selector

    const float* xrow = X + (size_t)(gw * 16 + m) * DD;

    // A fragments for all four 32-wide K chunks (hi + lo), in contiguous k order (K = h*16 + e), same as packed B.
    BF16x16 ahi[4], alo[4];
#pragma unroll
    for (int kc = 0; kc < 4; ++kc) {
        const float* p0 = xrow + kc * 32 + h * 16; // e in [0,8)  -> K = h*16 + e
        const float* p1 = p0 + 8;                  // e in [8,16) -> K = h*16 + e
        float f[16];
#pragma unroll
        for (int i = 0; i < 8; ++i) { f[i] = p0[i]; f[8 + i] = p1[i]; }
#pragma unroll
        for (int i = 0; i < 16; ++i) {
            float xv = f[i];
            if (relu) xv = fmaxf(xv, 0.0f);
            const unsigned short hi_ = f2bf(xv);
            ahi[kc].s[i] = hi_;
            alo[kc].s[i] = f2bf(xv - bf2f(hi_));
        }
    }

    for (int wi = 0; wi < 4; ++wi) {
        float* dst = (wi == 0) ? Ko : (wi == 1) ? Qo : (wi == 2) ? Vo : So;
        const unsigned short* bh0 = Bhi + wi * DD * DD;
        const unsigned short* bl0 = Blo + wi * DD * DD;
        for (int nt = 0; nt < 8; ++nt) {
            const float binit = (wi == 3) ? bias[nt * 16 + m] : 0.0f;
            v8f acc;
#pragma unroll
            for (int r = 0; r < 8; ++r) acc[r] = binit;
#pragma unroll
            for (int kc = 0; kc < 4; ++kc) {
                const int off = ((kc * 8 + nt) * 32 + lane) * 16;
                BF16x16 bh, bl;
                bh.u = *(const v8u*)(bh0 + off);
                bl.u = *(const v8u*)(bl0 + off);
                acc = __builtin_amdgcn_wmma_f32_16x16x32_bf16(
                    false, ahi[kc].v, false, bh.v, (short)0, acc, false, false);
                acc = __builtin_amdgcn_wmma_f32_16x16x32_bf16(
                    false, alo[kc].v, false, bh.v, (short)0, acc, false, false);
                acc = __builtin_amdgcn_wmma_f32_16x16x32_bf16(
                    false, ahi[kc].v, false, bl.v, (short)0, acc, false, false);
            }
            // C/D layout: VGPR r, lanes 0-15 -> M=r, lanes 16-31 -> M=8+r; N = lane&15
            const int row0 = gw * 16 + h * 8;
            const int n    = nt * 16 + m;
#pragma unroll
            for (int r = 0; r < 8; ++r) dst[(size_t)(row0 + r) * DD + n] = acc[r];
        }
    }
}

// ---------------------------------------------------------------------------
// Edge phase: one wave per edge. eta = sigmoid(k[tgt]+q[src]); msg = eta*v[src];
// scatter-add into out[tgt]. k/q/v (77MB) are L2-resident on MI455X (192MB L2).
// ---------------------------------------------------------------------------
__global__ __launch_bounds__(256)
void edge_msg(const int* __restrict__ ei,
              const float* __restrict__ K, const float* __restrict__ Q,
              const float* __restrict__ V, float* __restrict__ out) {
    const long long t = (long long)blockIdx.x * blockDim.x + threadIdx.x;
    const int e = (int)(t >> 5);
    if (e >= NE) return;
    const int lane = threadIdx.x & 31;
    const int src  = ei[e];          // edge_index[0] = src
    const int tgt  = ei[NE + e];     // edge_index[1] = tgt
    const int d    = lane * 4;

    const float4 kv = *(const float4*)(K + (size_t)tgt * DD + d);
    const float4 qv = *(const float4*)(Q + (size_t)src * DD + d);
    const float4 vv = *(const float4*)(V + (size_t)src * DD + d);

    float4 msg;
    msg.x = vv.x / (1.0f + __expf(-(kv.x + qv.x)));
    msg.y = vv.y / (1.0f + __expf(-(kv.y + qv.y)));
    msg.z = vv.z / (1.0f + __expf(-(kv.z + qv.z)));
    msg.w = vv.w / (1.0f + __expf(-(kv.w + qv.w)));

    float* o = out + (size_t)tgt * DD + d;
    (void)__hip_atomic_fetch_add(&o[0], msg.x, __ATOMIC_RELAXED, __HIP_MEMORY_SCOPE_AGENT);
    (void)__hip_atomic_fetch_add(&o[1], msg.y, __ATOMIC_RELAXED, __HIP_MEMORY_SCOPE_AGENT);
    (void)__hip_atomic_fetch_add(&o[2], msg.z, __ATOMIC_RELAXED, __HIP_MEMORY_SCOPE_AGENT);
    (void)__hip_atomic_fetch_add(&o[3], msg.w, __ATOMIC_RELAXED, __HIP_MEMORY_SCOPE_AGENT);
}

extern "C" void kernel_launch(void* const* d_in, const int* in_sizes, int n_in,
                              void* d_out, int out_size, void* d_ws, size_t ws_size,
                              hipStream_t stream) {
    (void)in_sizes; (void)n_in; (void)out_size; (void)ws_size;
    const float* x   = (const float*)d_in[0];
    const int*   ei  = (const int*)d_in[1];
    const float* Wk1 = (const float*)d_in[2];
    const float* Wq1 = (const float*)d_in[3];
    const float* Wv1 = (const float*)d_in[4];
    const float* Ws1 = (const float*)d_in[5];
    const float* b1  = (const float*)d_in[6];
    const float* Wk2 = (const float*)d_in[7];
    const float* Wq2 = (const float*)d_in[8];
    const float* Wv2 = (const float*)d_in[9];
    const float* Ws2 = (const float*)d_in[10];
    const float* b2  = (const float*)d_in[11];
    float* out = (float*)d_out;

    const size_t nd = (size_t)NN * DD;
    float* Kb = (float*)d_ws;
    float* Qb = Kb + nd;
    float* Vb = Qb + nd;
    float* Hb = Vb + nd;
    unsigned short* Whi = (unsigned short*)(Hb + nd);
    unsigned short* Wlo = Whi + 4 * DD * DD;

    const dim3 blk(256);
    const int gemm_blocks = (MTILES + 7) / 8;              // 8 waves/block, 1 M-tile/wave
    const int edge_blocks = (NE * 32 + 255) / 256;         // 1 wave/edge

    // ---- layer 1 ----
    pack_weights<<<4, blk, 0, stream>>>(Wk1, Wq1, Wv1, Ws1, Whi, Wlo);
    gemm_kqvs<<<gemm_blocks, blk, 0, stream>>>(x, Whi, Wlo, b1, Kb, Qb, Vb, Hb, 0);
    edge_msg<<<edge_blocks, blk, 0, stream>>>(ei, Kb, Qb, Vb, Hb);

    // ---- layer 2 (ReLU fused into A-load of the GEMM) ----
    pack_weights<<<4, blk, 0, stream>>>(Wk2, Wq2, Wv2, Ws2, Whi, Wlo);
    gemm_kqvs<<<gemm_blocks, blk, 0, stream>>>(Hb, Whi, Wlo, b2, Kb, Qb, Vb, out, 1);
    edge_msg<<<edge_blocks, blk, 0, stream>>>(ei, Kb, Qb, Vb, out);
}
